// HiddenToEvents_75797582839976
// MI455X (gfx1250) — matrix operand order repaired and checked by reading, not compiled, not run on hardware
//
#include <hip/hip_runtime.h>
#include <hip/hip_bf16.h>
#include <math.h>

// ---------------------------------------------------------------------------
// Problem constants (from reference): B=32, S=512, H=2048, FP=SP=1024
// N = 16384 tokens. Outputs: log_prob [N] then prob_all_mat [N, 2050].
// ---------------------------------------------------------------------------
#define NTOK   16384
#define HDIM   2048
#define FPP    1024
#define SPP    1024
#define OUTC   (FPP + SPP + 2)   // 2050

#define KC     512               // K-chunk staged in LDS per TDM transfer
#define NCH    (HDIM / KC)       // 4 chunks
#define AROWE  (KC + 8)          // padded LDS row stride in elements (1040 B)

typedef __attribute__((ext_vector_type(16))) __bf16 v16bf;
typedef __attribute__((ext_vector_type(8)))  __bf16 v8bf;
typedef __attribute__((ext_vector_type(8)))  float  v8f;
typedef __attribute__((ext_vector_type(4)))  float  f4;
typedef __attribute__((ext_vector_type(4)))  unsigned short us4;
typedef __attribute__((ext_vector_type(4)))  unsigned int   v4u;
typedef __attribute__((ext_vector_type(8)))  int            v8i;
typedef __attribute__((ext_vector_type(4)))  int            v4i;

union AFrag { v16bf v; v8bf h[2]; };

#if defined(__has_builtin)
#  if __has_builtin(__builtin_amdgcn_tensor_load_to_lds)
#    define HAVE_TDM 1
#  endif
#endif

// fp32 -> bf16 (round to nearest even), bit-level
__device__ __forceinline__ unsigned short f32_to_bf16(float f) {
    unsigned u = __float_as_uint(f);
    u += 0x7FFFu + ((u >> 16) & 1u);
    return (unsigned short)(u >> 16);
}

__global__ void cvt_bf16_kernel(const f4* __restrict__ src,
                                us4* __restrict__ dst, int n4) {
    int i = blockIdx.x * blockDim.x + threadIdx.x;
    if (i < n4) {
        f4 v = src[i];
        us4 o;
        o[0] = f32_to_bf16(v[0]); o[1] = f32_to_bf16(v[1]);
        o[2] = f32_to_bf16(v[2]); o[3] = f32_to_bf16(v[3]);
        dst[i] = o;
    }
}

#ifdef HAVE_TDM
// Issue one TDM 2D-tile load: 16 rows x KC bf16 elems, global stride HDIM,
// LDS rows padded by 16B (pad_interval=7 -> every 256 DWORDs, pad_amount=3
// -> 4 DWORDs) so the 16 A-rows land on disjoint bank groups.
// This toolchain exposes the 6-arg builtin:
//   (uint32x4 g0, int32x8 g1, int32x4 g2, int32x4 g3, int32x8 g4, i32 cpol)
__device__ __forceinline__ void tdm_load_A(const unsigned short* gptr,
                                           unsigned lds_off) {
    unsigned long long ga = (unsigned long long)(uintptr_t)gptr;
    v4u g0;
    g0[0] = 1u;                                   // count=1, user descriptor
    g0[1] = lds_off;                              // lds_addr (bytes)
    g0[2] = (unsigned)ga;                         // global_addr[31:0]
    g0[3] = (unsigned)((ga >> 32) & 0x01FFFFFFu)  // global_addr[56:32]
            | 0x80000000u;                        // type=2 ("image")
    v8i g1;
    g1[0] = (1 << 16)        // data_size = 2 bytes
          | (1 << 20)        // pad_enable
          | (7 << 22)        // pad_interval = 256 DWORDs (one row)
          | (3 << 25);       // pad_amount   = 4 DWORDs (16 B)
    g1[1] = KC << 16;        // tensor_dim0[15:0]
    g1[2] = 16 << 16;        // tensor_dim1[15:0]
    g1[3] = KC << 16;        // tile_dim0
    g1[4] = 16;              // tile_dim1 (tile_dim2 = 0)
    g1[5] = HDIM;            // tensor_dim0_stride[31:0]
    g1[6] = 0;
    g1[7] = 0;
    v4i z4 = (v4i){0, 0, 0, 0};                   // 2D tensor: groups 2/3 unused
    v8i z8 = (v8i){0, 0, 0, 0, 0, 0, 0, 0};
    __builtin_amdgcn_tensor_load_to_lds(g0, g1, z4, z4, z8, 0);
}
#endif

// ---------------------------------------------------------------------------
// Fused GEMM (bf16 WMMA, f32 accum, TDM-staged A) + dual softmax +
// end-head sigmoid + masked scatter.
// Grid: 1024 blocks of 16 tokens.  Block: 512 threads (16 wave32).
// Wave w covers columns [w*128, w*128+128); waves 0..7 = hcw, 8..15 = roo.
// ---------------------------------------------------------------------------
__global__ void __launch_bounds__(512)
hidden_to_events_kernel(const float* __restrict__ X,      // [N, H] fp32
                        const int*   __restrict__ pY,     // [N]
                        const int*   __restrict__ Y,      // [N]
                        const float* __restrict__ W_end,  // [H]
                        const float* __restrict__ b_end,  // [1]
                        const float* __restrict__ b_hcw,  // [FP]
                        const float* __restrict__ b_roo,  // [SP]
                        const unsigned short* __restrict__ Xb,  // [N, H] bf16
                        const unsigned short* __restrict__ Wb,  // [2048, H] bf16
                        float* __restrict__ out_log,      // [N]
                        float* __restrict__ out_prob)     // [N, 2050]
{
    __shared__ unsigned short Asmem[2][16][AROWE];   // double-buffered A tile
    __shared__ float red[2][16][8];
    __shared__ float maxv[2][16];
    __shared__ float sums[2][16];
    __shared__ float nend[16];
    __shared__ int   spY[16];
    __shared__ int   sYv[16];

    const int tid  = threadIdx.x;
    const int wave = tid >> 5;         // 0..15
    const int lane = tid & 31;
    const int half = lane >> 4;        // 0/1
    const int l16  = lane & 15;
    const int h    = wave >> 3;        // 0 = hcw head, 1 = roo head
    const int w8   = wave & 7;         // wave index within head
    const int m0   = blockIdx.x * 16;  // first token of this block

    if (tid < 16) {
        spY[tid] = pY[m0 + tid];
        sYv[tid] = Y[m0 + tid];
    }

#ifdef HAVE_TDM
    const unsigned ldsA0 = (unsigned)(uintptr_t)(&Asmem[0][0][0]);
    const unsigned ldsA1 = (unsigned)(uintptr_t)(&Asmem[1][0][0]);
    if (wave == 0) {   // stage chunk 0
        tdm_load_A(Xb + (size_t)m0 * HDIM, ldsA0);
        __builtin_amdgcn_s_wait_tensorcnt(0);
    }
#else
    {   // cooperative fallback copy of chunk 0 (512 thr x 16 elems)
        int e = tid * 16;
        int r = e >> 9, c = e & (KC - 1);
        const unsigned short* src = Xb + (size_t)(m0 + r) * HDIM + c;
        for (int j = 0; j < 16; ++j) Asmem[0][r][c + j] = src[j];
    }
#endif
    __syncthreads();

    // ---------------- end head: wave w -> token m0+w (fp32 GEMV) ----------
    {
        const int tok = m0 + wave;
        const float* xr = X + (size_t)tok * HDIM;
        float s = 0.f;
        for (int c = lane; c < HDIM; c += 32) s += xr[c] * W_end[c];
        #pragma unroll
        for (int off = 16; off; off >>= 1) s += __shfl_xor(s, off, 32);
        float ep = 1.f / (1.f + __expf(-(s + b_end[0])));
        if (lane == 0) {
            nend[wave] = 1.f - ep;
            int py = spY[wave];
            float c01 = (py == 0) ? ep : 0.f;
            size_t rb = (size_t)tok * OUTC;
            out_prob[rb + 0] = c01;
            out_prob[rb + 1] = c01;
            if (py == 0)      out_log[tok] = __logf(ep);
            else if (py > 2)  out_log[tok] = 0.f;   // reference default prob=1
        }
    }

    // ---------------- GEMM: 16 tokens x 128 cols per wave, K = 2048 -------
    v8f acc[8];
    #pragma unroll
    for (int t = 0; t < 8; ++t) acc[t] = (v8f){0.f,0.f,0.f,0.f,0.f,0.f,0.f,0.f};

    const int n0 = wave * 128;                       // global column base
    const unsigned short* brow[8];
    #pragma unroll
    for (int t = 0; t < 8; ++t)
        brow[t] = Wb + (size_t)(n0 + t * 16 + l16) * HDIM + half * 16;

    for (int c = 0; c < NCH; ++c) {
        const int cb = c & 1;
#ifdef HAVE_TDM
        // prefetch next chunk via TDM while computing this one
        if (wave == 0 && c + 1 < NCH)
            tdm_load_A(Xb + (size_t)m0 * HDIM + (c + 1) * KC, (c & 1) ? ldsA0 : ldsA1);
#endif
        const int kg = c * KC;
        for (int kk = 0; kk < KC; kk += 32) {
            // A fragment from LDS (contiguous k layout: K = 16h + s, matching B)
            AFrag a;
            a.h[0] = *(const v8bf*)(&Asmem[cb][l16][kk + half * 16]);
            a.h[1] = *(const v8bf*)(&Asmem[cb][l16][kk + half * 16 + 8]);
            #pragma unroll
            for (int t = 0; t < 8; ++t) {
                v16bf bv = *(const v16bf*)(brow[t] + kg + kk);   // 32B contiguous
                acc[t] = __builtin_amdgcn_wmma_f32_16x16x32_bf16(
                             false, a.v, false, bv, (short)0, acc[t], false, false);
            }
        }
#ifdef HAVE_TDM
        if (wave == 0 && c + 1 < NCH) __builtin_amdgcn_s_wait_tensorcnt(0);
        __syncthreads();
#else
        __syncthreads();
        if (c + 1 < NCH) {
            int e = tid * 16;
            int r = e >> 9, cc = e & (KC - 1);
            const unsigned short* src =
                Xb + (size_t)(m0 + r) * HDIM + (c + 1) * KC + cc;
            for (int j = 0; j < 16; ++j) Asmem[(c + 1) & 1][r][cc + j] = src[j];
            __syncthreads();
        }
#endif
    }

    // Bias add (column biases; zero in reference but kept for fidelity)
    #pragma unroll
    for (int t = 0; t < 8; ++t) {
        int colh = w8 * 128 + t * 16 + l16;          // column within head
        float bb = h ? b_roo[colh] : b_hcw[colh];
        #pragma unroll
        for (int r = 0; r < 8; ++r) acc[t][r] += bb;
    }

    // ---------------- softmax: block max per (head, token) ----------------
    #pragma unroll
    for (int r = 0; r < 8; ++r) {
        float m = acc[0][r];
        #pragma unroll
        for (int t = 1; t < 8; ++t) m = fmaxf(m, acc[t][r]);
        #pragma unroll
        for (int off = 1; off < 16; off <<= 1) m = fmaxf(m, __shfl_xor(m, off, 32));
        if (l16 == 0) red[h][r + 8 * half][w8] = m;
    }
    __syncthreads();
    if (tid < 32) {
        int hh = tid >> 4, tt = tid & 15;
        float m = red[hh][tt][0];
        #pragma unroll
        for (int w = 1; w < 8; ++w) m = fmaxf(m, red[hh][tt][w]);
        maxv[hh][tt] = m;
    }
    __syncthreads();

    // ---------------- softmax: block sum-exp per (head, token) ------------
    #pragma unroll
    for (int r = 0; r < 8; ++r) {
        int tt = r + 8 * half;
        float mv = maxv[h][tt];
        float s = 0.f;
        #pragma unroll
        for (int t = 0; t < 8; ++t) s += __expf(acc[t][r] - mv);
        #pragma unroll
        for (int off = 1; off < 16; off <<= 1) s += __shfl_xor(s, off, 32);
        if (l16 == 0) red[h][tt][w8] = s;
    }
    __syncthreads();
    if (tid < 32) {
        int hh = tid >> 4, tt = tid & 15;
        float s = red[hh][tt][0];
        #pragma unroll
        for (int w = 1; w < 8; ++w) s += red[hh][tt][w];
        sums[hh][tt] = s;
    }
    __syncthreads();

    // ---------------- masked scatter + gathered log_prob ------------------
    #pragma unroll
    for (int t = 0; t < 8; ++t) {
        const int colh = w8 * 128 + t * 16 + l16;        // col within head
        const size_t gcol = 2 + (size_t)h * FPP + colh;  // col in prob_all_mat
        #pragma unroll
        for (int r = 0; r < 8; ++r) {
            const int tt  = r + 8 * half;
            const int tok = m0 + tt;
            const int py  = spY[tt];
            float p  = __expf(acc[t][r] - maxv[h][tt]) / sums[h][tt];
            float pn = p * nend[tt];
            bool sel = (py == h + 1);
            out_prob[(size_t)tok * OUTC + gcol] = sel ? pn : 0.f;
            if (sel) {
                int Yv  = sYv[tt];
                int idx = h ? min(max(Yv - 2 - FPP, 0), SPP - 1)
                            : min(max(Yv - 2,       0), FPP - 1);
                if (idx == colh) out_log[tok] = __logf(pn);
            }
        }
    }
}

// ---------------------------------------------------------------------------
extern "C" void kernel_launch(void* const* d_in, const int* in_sizes, int n_in,
                              void* d_out, int out_size, void* d_ws, size_t ws_size,
                              hipStream_t stream) {
    const float* X     = (const float*)d_in[0];
    const int*   pY    = (const int*)  d_in[1];
    const int*   Yv    = (const int*)  d_in[2];
    const float* W_end = (const float*)d_in[3];
    const float* b_end = (const float*)d_in[4];
    const float* W_hcw = (const float*)d_in[5];
    const float* b_hcw = (const float*)d_in[6];
    const float* W_roo = (const float*)d_in[7];
    const float* b_roo = (const float*)d_in[8];

    float* out_log  = (float*)d_out;
    float* out_prob = out_log + NTOK;

    unsigned short* Xb = (unsigned short*)d_ws;                 // 64 MiB
    unsigned short* Wb = Xb + (size_t)NTOK * HDIM;              // 8 MiB

    const int nX4 = NTOK * HDIM / 4;
    const int nW4 = FPP * HDIM / 4;
    cvt_bf16_kernel<<<(nX4 + 255) / 256, 256, 0, stream>>>(
        (const f4*)X, (us4*)Xb, nX4);
    cvt_bf16_kernel<<<(nW4 + 255) / 256, 256, 0, stream>>>(
        (const f4*)W_hcw, (us4*)Wb, nW4);
    cvt_bf16_kernel<<<(nW4 + 255) / 256, 256, 0, stream>>>(
        (const f4*)W_roo, (us4*)(Wb + (size_t)FPP * HDIM), nW4);

    hidden_to_events_kernel<<<NTOK / 16, 512, 0, stream>>>(
        X, pY, Yv, W_end, b_end, b_hcw, b_roo, Xb, Wb, out_log, out_prob);
}
